// TimeAwareCrossAttention_84731114815576
// MI455X (gfx1250) — compile-verified, matrix operand order checked by reading, not run on hardware
//
#include <hip/hip_runtime.h>
#include <hip/hip_bf16.h>

typedef __attribute__((ext_vector_type(16))) _Float16 v16h;
typedef __attribute__((ext_vector_type(8)))  _Float16 v8h;
typedef __attribute__((ext_vector_type(8)))  float    v8f;
typedef unsigned int u32x4 __attribute__((ext_vector_type(4)));
typedef int          i32x8 __attribute__((ext_vector_type(8)));
typedef int          i32x4 __attribute__((ext_vector_type(4)));

static __device__ __forceinline__ v8h ld8(const _Float16* p) {
  return *(const v8h*)p;
}
static __device__ __forceinline__ v16h mk16(v8h a, v8h b) {
  return __builtin_shufflevector(a, b, 0,1,2,3,4,5,6,7,8,9,10,11,12,13,14,15);
}
// One 16-bit A/B WMMA fragment from a k-major row. ISA layout: lanes 0-15 hold
// k = kk*32 + {0..7, 16..23}, lanes 16-31 hold k = kk*32 + {8..15, 24..31};
// each half is a contiguous 16B chunk -> 2 x b128.
static __device__ __forceinline__ v16h fragAB(const _Float16* rowp, int kk, int hi8) {
  return mk16(ld8(rowp + kk*32 + hi8), ld8(rowp + kk*32 + 16 + hi8));
}
static __device__ __forceinline__ v8f wmma_f16(v16h a, v16h b, v8f c) {
  return __builtin_amdgcn_wmma_f32_16x16x32_f16(false, a, false, b, (short)0, c,
                                                false, false);
}
static __device__ __forceinline__ float tdecay(float tq, float tk) {
  float dt = fabsf(tq - tk);
  dt = fminf(dt, 1000.0f);
  return __expf(dt * -0.01f);
}

// ---------------------------------------------------------------------------
// Tensor Data Mover: DMA a 2D tile (tile_d0 elements x tile_d1 rows, 2-byte
// elements, row stride stride0 elements) from global into LDS at lds_off.
// D# bit layout per CDNA5 ISA 8.3/8.4. tile_d1 == 0 -> 1D tile.
// ---------------------------------------------------------------------------
static __device__ __forceinline__ void tdm_load_2d(
    unsigned int lds_off, const void* gaddr,
    unsigned int tile_d0, unsigned int tile_d1,
    unsigned int tensor_d0, unsigned int tensor_d1, unsigned long long stride0)
{
  unsigned long long ga = (unsigned long long)(uintptr_t)gaddr;
  u32x4 g0;
  g0[0] = 1u;                                   // count=1, user descriptor
  g0[1] = lds_off;                              // lds_addr (bytes)
  g0[2] = (unsigned int)ga;                     // global_addr[31:0]
  g0[3] = (unsigned int)(ga >> 32) | (2u << 30);// global_addr[56:32] | type=2
  i32x8 g1;
  g1[0] = (int)(1u << 16);                      // data_size=1 (2 bytes)
  g1[1] = (int)((tensor_d0 & 0xFFFFu) << 16);   // [47:32]=0, [63:48]=dim0 lo
  g1[2] = (int)(((tensor_d0 >> 16) & 0xFFFFu) | ((tensor_d1 & 0xFFFFu) << 16));
  g1[3] = (int)(((tensor_d1 >> 16) & 0xFFFFu) | ((tile_d0 & 0xFFFFu) << 16));
  g1[4] = (int)(tile_d1 & 0xFFFFu);             // tile_dim1; tile_dim2=0
  g1[5] = (int)(unsigned int)stride0;           // dim0_stride[31:0]
  g1[6] = (int)((unsigned int)(stride0 >> 32) & 0xFFFFu); // dim0_stride[47:32]
  g1[7] = 0;
  i32x4 gz = {0, 0, 0, 0};
#if __clang_major__ >= 23
  i32x8 gz8 = {0, 0, 0, 0, 0, 0, 0, 0};
  __builtin_amdgcn_tensor_load_to_lds(g0, g1, gz, gz, gz8, 0);
#else
  __builtin_amdgcn_tensor_load_to_lds(g0, g1, gz, gz, 0);
#endif
}

// ---------------------------------------------------------------------------
// C[m][n] = sum_k X[m][k] * W[n][k] + bias[n]   (i.e. X @ W^T + b), K = 1024.
// Workgroup: 256 threads / 8 waves; tile 128(M) x 64(N); each wave 32x32.
// out_mode 0: f16 to [B, H, L, 64]       (row-major per head)
// out_mode 1: f32 flat [M, 1024]         (final output)
// out_mode 2: f16 to [B, H, 64, L]       (head-dim-major, for V -> TDM tiles)
// ---------------------------------------------------------------------------
template <typename TIN>
__global__ void __launch_bounds__(256) proj_gemm(
    const TIN* __restrict__ X, const float* __restrict__ W,
    const float* __restrict__ bias, void* __restrict__ OUT,
    int L, int out_mode)
{
  __shared__ __align__(16) _Float16 xbuf[128 * 32];
  __shared__ __align__(16) _Float16 wbuf[64 * 32];

  const int tid  = threadIdx.x;
  const int lane = tid & 31, wave = tid >> 5;
  const int l16  = lane & 15, hi8 = (lane >> 4) << 3;
  const int m0 = blockIdx.x * 128, n0 = blockIdx.y * 64;
  const int wm = (wave >> 1) * 32, wn = (wave & 1) * 32;

  v8f zf = {0.f,0.f,0.f,0.f,0.f,0.f,0.f,0.f};
  v8f c00 = zf, c01 = zf, c10 = zf, c11 = zf;

  for (int kc = 0; kc < 1024; kc += 32) {
    __syncthreads();
#pragma unroll
    for (int e = 0; e < 16; ++e) {          // 128x32 X chunk -> f16 LDS
      int idx = tid * 16 + e;
      int row = idx >> 5, kk = idx & 31;
      xbuf[idx] = (_Float16)(float)X[(size_t)(m0 + row) * 1024 + kc + kk];
    }
#pragma unroll
    for (int e = 0; e < 8; ++e) {           // 64x32 W chunk -> f16 LDS
      int idx = tid * 8 + e;
      int row = idx >> 5, kk = idx & 31;
      wbuf[idx] = (_Float16)W[(size_t)(n0 + row) * 1024 + kc + kk];
    }
    __syncthreads();

    v16h a0f = fragAB(xbuf + (wm + l16) * 32,      0, hi8);
    v16h a1f = fragAB(xbuf + (wm + 16 + l16) * 32, 0, hi8);
    v16h b0f = fragAB(wbuf + (wn + l16) * 32,      0, hi8);
    v16h b1f = fragAB(wbuf + (wn + 16 + l16) * 32, 0, hi8);
    c00 = wmma_f16(a0f, b0f, c00);
    c01 = wmma_f16(a0f, b1f, c01);
    c10 = wmma_f16(a1f, b0f, c10);
    c11 = wmma_f16(a1f, b1f, c11);
  }

  v8f cc[4] = {c00, c01, c10, c11};
#pragma unroll
  for (int t = 0; t < 4; ++t) {
    int mi = t >> 1, ni = t & 1;
#pragma unroll
    for (int r = 0; r < 8; ++r) {
      int gm = m0 + wm + mi * 16 + r + hi8;     // C layout: VGPR r -> M = r (+8 hi half)
      int gn = n0 + wn + ni * 16 + l16;         //           N = lane % 16
      float v = cc[t][r] + bias[gn];
      if (out_mode == 0) {
        int bb = gm / L, ll = gm - bb * L;
        int hh = gn >> 6, hd = gn & 63;
        ((_Float16*)OUT)[((((size_t)bb * 16 + hh) * (size_t)L + ll) << 6) + hd] =
            (_Float16)v;
      } else if (out_mode == 1) {
        ((float*)OUT)[(size_t)gm * 1024 + gn] = v;
      } else {
        int bb = gm / L, ll = gm - bb * L;
        int hh = gn >> 6, hd = gn & 63;
        ((_Float16*)OUT)[(((size_t)bb * 16 + hh) * 64 + hd) * (size_t)L + ll] =
            (_Float16)v;
      }
    }
  }
}

// ---------------------------------------------------------------------------
// Fused time-decayed flash attention. Grid: B*H*(L1/128) blocks, 256 threads.
// Each wave owns a 16-row query tile. K/V chunks (32 keys) are DMA'd into
// double-buffered LDS by the Tensor Data Mover (wave 0 issues, TENSORcnt
// synchronized), overlapping the next chunk's DMA with WMMA compute.
// Per chunk: 4 WMMAs for S (K=64), decay + online softmax, 4 WMMAs for P@V.
// ---------------------------------------------------------------------------
__global__ void __launch_bounds__(256) attn_fused(
    const _Float16* __restrict__ Qw, const _Float16* __restrict__ Kw,
    const _Float16* __restrict__ Vt, const int* __restrict__ qts,
    const int* __restrict__ kts, _Float16* __restrict__ Aout)
{
  const int L1 = 1024, L2 = 2048, NH = 16;
  __shared__ __align__(16) _Float16 kbuf[2][32 * 64];  // [key][dim]
  __shared__ __align__(16) _Float16 vbuf[2][64 * 32];  // [dim][key] (TDM 2D tile)
  __shared__ __align__(16) _Float16 pbuf[8][16 * 32];  // per-wave P tile

  const int wg   = blockIdx.x;
  const int qblk = wg & 7, h = (wg >> 3) & 15, b = wg >> 7;
  const int tid  = threadIdx.x, lane = tid & 31, wave = tid >> 5;
  const int l16  = lane & 15, hi8 = (lane >> 4) << 3;
  const int qbase = qblk * 128 + wave * 16;

  const _Float16* Kbase  = Kw + (size_t)(b * NH + h) * L2 * 64;   // [L2][64]
  const _Float16* Vtbase = Vt + (size_t)(b * NH + h) * 64 * L2;   // [64][L2]

  // Q tile for this wave stays in registers: 2 A-fragments (k=0..31, 32..63)
  const _Float16* Qrow = Qw + ((size_t)((b * NH + h) * L1) + qbase + l16) * 64;
  v16h qf0 = fragAB(Qrow, 0, hi8);
  v16h qf1 = fragAB(Qrow, 1, hi8);

  float tqv[8];
#pragma unroll
  for (int r = 0; r < 8; ++r)
    tqv[r] = (float)qts[b * L1 + qbase + r + hi8];

  float mrow[8], lrow[8];
#pragma unroll
  for (int r = 0; r < 8; ++r) { mrow[r] = -3.0e30f; lrow[r] = 0.f; }
  v8f zf = {0.f,0.f,0.f,0.f,0.f,0.f,0.f,0.f};
  v8f of0 = zf, of1 = zf, of2 = zf, of3 = zf;

  const int* ktb = kts + b * L2;
  _Float16* pw = pbuf[wave];

  auto issue_chunk = [&](int chunk, int bufp) {
    // K chunk: contiguous 32*64 halfs (1D tile)
    tdm_load_2d((unsigned int)(uintptr_t)(void*)kbuf[bufp],
                Kbase + (size_t)chunk * 32 * 64,
                /*tile_d0=*/2048, /*tile_d1=*/0,
                /*tensor_d0=*/(unsigned int)(L2 * 64), /*tensor_d1=*/1,
                /*stride0=*/2048ull);
    // V chunk: 2D tile, 32 keys x 64 dim-rows, row stride L2 -> lands [dim][32]
    tdm_load_2d((unsigned int)(uintptr_t)(void*)vbuf[bufp],
                Vtbase + (size_t)chunk * 32,
                /*tile_d0=*/32, /*tile_d1=*/64,
                /*tensor_d0=*/(unsigned int)L2, /*tensor_d1=*/64,
                /*stride0=*/(unsigned long long)L2);
  };

  const int NCH = L2 / 32;   // 64 chunks
  if (wave == 0) issue_chunk(0, 0);

  for (int i = 0; i < NCH; ++i) {
    const int cur = i & 1;
    __syncthreads();   // all waves done reading the buffer about to be refilled
    if (wave == 0) {
      if (i + 1 < NCH) {
        issue_chunk(i + 1, (i + 1) & 1);
        // the 2 ops just issued may remain outstanding; chunk i must be done
        __builtin_amdgcn_s_wait_tensorcnt(2);
      } else {
        __builtin_amdgcn_s_wait_tensorcnt(0);
      }
    }
    __syncthreads();   // chunk i visible to all waves; DMA of i+1 overlaps below

    const _Float16* kb = kbuf[cur];
    const _Float16* vb = vbuf[cur];
    const int kv0 = i * 32;

    // S = Q(16x64) @ K^T(64x32): two 16-key N-subtiles, k reduced 2x32
    v8f s0 = zf, s1 = zf;
    {
      const _Float16* kr0 = kb + l16 * 64;
      const _Float16* kr1 = kb + (16 + l16) * 64;
      s0 = wmma_f16(qf0, fragAB(kr0, 0, hi8), s0);
      s0 = wmma_f16(qf1, fragAB(kr0, 1, hi8), s0);
      s1 = wmma_f16(qf0, fragAB(kr1, 0, hi8), s1);
      s1 = wmma_f16(qf1, fragAB(kr1, 1, hi8), s1);
    }

    float tk0 = (float)ktb[kv0 + l16];
    float tk1 = (float)ktb[kv0 + 16 + l16];

    float p0[8], p1[8];
#pragma unroll
    for (int r = 0; r < 8; ++r) {
      float sc0 = s0[r] * 0.125f * tdecay(tqv[r], tk0);
      float sc1 = s1[r] * 0.125f * tdecay(tqv[r], tk1);
      // row max across the 16 columns held by lanes (xor<16 stays in half-wave)
      float mx = fmaxf(sc0, sc1);
      mx = fmaxf(mx, __shfl_xor(mx, 1, 32));
      mx = fmaxf(mx, __shfl_xor(mx, 2, 32));
      mx = fmaxf(mx, __shfl_xor(mx, 4, 32));
      mx = fmaxf(mx, __shfl_xor(mx, 8, 32));
      float mnew = fmaxf(mrow[r], mx);
      float ex0 = __expf(sc0 - mnew), ex1 = __expf(sc1 - mnew);
      float rs = ex0 + ex1;
      rs += __shfl_xor(rs, 1, 32);
      rs += __shfl_xor(rs, 2, 32);
      rs += __shfl_xor(rs, 4, 32);
      rs += __shfl_xor(rs, 8, 32);
      float alpha = __expf(mrow[r] - mnew);
      lrow[r] = lrow[r] * alpha + rs;
      mrow[r] = mnew;
      of0[r] *= alpha; of1[r] *= alpha; of2[r] *= alpha; of3[r] *= alpha;
      p0[r] = ex0; p1[r] = ex1;
    }

    // C-layout -> A-layout transpose of P through per-wave LDS tile
#pragma unroll
    for (int r = 0; r < 8; ++r) {
      pw[(r + hi8) * 32 + l16]      = (_Float16)p0[r];
      pw[(r + hi8) * 32 + 16 + l16] = (_Float16)p1[r];
    }
    asm volatile("s_wait_dscnt 0" ::: "memory");
    v16h pf = fragAB(pw + l16 * 32, 0, hi8);

    // O += P(16x32) @ V(32x64), 4 N-subtiles of 16 dims
    of0 = wmma_f16(pf, fragAB(vb + (0 * 16 + l16) * 32, 0, hi8), of0);
    of1 = wmma_f16(pf, fragAB(vb + (1 * 16 + l16) * 32, 0, hi8), of1);
    of2 = wmma_f16(pf, fragAB(vb + (2 * 16 + l16) * 32, 0, hi8), of2);
    of3 = wmma_f16(pf, fragAB(vb + (3 * 16 + l16) * 32, 0, hi8), of3);
  }

  v8f oo[4] = {of0, of1, of2, of3};
#pragma unroll
  for (int dt = 0; dt < 4; ++dt) {
#pragma unroll
    for (int r = 0; r < 8; ++r) {
      int q = qbase + r + hi8;
      float v = oo[dt][r] / lrow[r];
      Aout[((size_t)(b * L1 + q)) * 1024 + h * 64 + dt * 16 + l16] = (_Float16)v;
    }
  }
}

// ---------------------------------------------------------------------------
extern "C" void kernel_launch(void* const* d_in, const int* in_sizes, int n_in,
                              void* d_out, int out_size, void* d_ws, size_t ws_size,
                              hipStream_t stream) {
  (void)in_sizes; (void)n_in; (void)out_size; (void)ws_size;
  const float* query = (const float*)d_in[0];
  const float* keyb  = (const float*)d_in[1];
  const float* value = (const float*)d_in[2];
  const int*   qts   = (const int*)d_in[3];
  const int*   kts   = (const int*)d_in[4];
  const float* Wq = (const float*)d_in[5];
  const float* bq = (const float*)d_in[6];
  const float* Wk = (const float*)d_in[7];
  const float* bk = (const float*)d_in[8];
  const float* Wv = (const float*)d_in[9];
  const float* bv = (const float*)d_in[10];
  const float* Wo = (const float*)d_in[11];
  const float* bo = (const float*)d_in[12];

  // Workspace (f16): Q 4MB, K 8MB, V^T 8MB, attn-out 4MB  (24MB total)
  _Float16* Qws  = (_Float16*)d_ws;
  _Float16* Kws  = Qws + (size_t)2 * 16 * 1024 * 64;
  _Float16* Vtws = Kws + (size_t)2 * 16 * 2048 * 64;
  _Float16* Aout = Vtws + (size_t)2 * 16 * 2048 * 64;

  dim3 blk(256);
  // Q/K projections (f32 in, f16 head-major out); V transposed for TDM tiles
  proj_gemm<float><<<dim3(2048 / 128, 16), blk, 0, stream>>>(query, Wq, bq, Qws, 1024, 0);
  proj_gemm<float><<<dim3(4096 / 128, 16), blk, 0, stream>>>(keyb,  Wk, bk, Kws, 2048, 0);
  proj_gemm<float><<<dim3(4096 / 128, 16), blk, 0, stream>>>(value, Wv, bv, Vtws, 2048, 2);
  // Fused time-decayed flash attention: B*H*(L1/128) = 256 workgroups
  attn_fused<<<dim3(256), blk, 0, stream>>>(Qws, Kws, Vtws, qts, kts, Aout);
  // Output projection (f16 in, f32 flat out to d_out)
  proj_gemm<_Float16><<<dim3(2048 / 128, 16), blk, 0, stream>>>(Aout, Wo, bo,
                                                                d_out, 1024, 1);
}
